// BarycentricMapping_89446988906808
// MI455X (gfx1250) — matrix operand order repaired and checked by reading, not compiled, not run on hardware
//
#include <hip/hip_runtime.h>
#include <hip/hip_bf16.h>

typedef __attribute__((ext_vector_type(16))) _Float16 v16h;
typedef __attribute__((ext_vector_type(8)))  _Float16 v8h;
typedef __attribute__((ext_vector_type(8)))  float    v8f;

#define N_PTS 8192
#define DIM   512
#define LOG2E 1.4426950408889634f
#define KSCALE (LOG2E / 0.05f)      /* -C/reg in base-2 domain */
#define NEG_BIG (-3.0e38f)

__device__ __forceinline__ float fexp2(float x) {
#if __has_builtin(__builtin_amdgcn_exp2f)
  return __builtin_amdgcn_exp2f(x);
#else
  return exp2f(x);
#endif
}
__device__ __forceinline__ float flog2(float x) {
#if __has_builtin(__builtin_amdgcn_logf)
  return __builtin_amdgcn_logf(x);   /* v_log_f32 is log2 */
#else
  return log2f(x);
#endif
}

/* online-LSE: fold a group of 4 values into (m,s) with a single rescale */
__device__ __forceinline__ void lse4(float& m, float& s,
                                     float x0, float x1, float x2, float x3) {
  float mx = fmaxf(fmaxf(x0, x1), fmaxf(x2, x3));
  float M  = fmaxf(m, mx);
  s = s * fexp2(m - M) + (fexp2(x0 - M) + fexp2(x1 - M)) +
                         (fexp2(x2 - M) + fexp2(x3 - M));
  m = M;
}

/* ---- convert one f32 matrix row-block to f16 + squared row norms ---- */
__global__ void k_prep(const float* __restrict__ X, _Float16* __restrict__ Xh,
                       float* __restrict__ n2) {
  int row = blockIdx.x, tid = threadIdx.x;
  const float* xr = X + (size_t)row * DIM;
  float s = 0.f;
  for (int c = tid; c < DIM; c += 256) {
    float v = xr[c];
    Xh[(size_t)row * DIM + c] = (_Float16)v;
    s += v * v;
  }
  __shared__ float red[256];
  red[tid] = s; __syncthreads();
  for (int off = 128; off > 0; off >>= 1) {
    if (tid < off) red[tid] += red[tid + off];
    __syncthreads();
  }
  if (tid == 0) n2[row] = red[0];
}

/* ---- LDS tile transpose XQh[8192][512] -> XQt[512][8192] (f16) ---- */
__global__ void k_transpose(const _Float16* __restrict__ XQh, _Float16* __restrict__ XQt) {
  __shared__ _Float16 tile[32][33];
  int j0 = blockIdx.x * 32, d0 = blockIdx.y * 32;
  int tx = threadIdx.x & 31, ty = threadIdx.x >> 5;   // ty: 0..7
#pragma unroll
  for (int r = 0; r < 4; r++) {
    int jj = ty + r * 8;
    tile[jj][tx] = XQh[(size_t)(j0 + jj) * DIM + d0 + tx];
  }
  __syncthreads();
#pragma unroll
  for (int r = 0; r < 4; r++) {
    int dd = ty + r * 8;
    XQt[(size_t)(d0 + dd) * N_PTS + j0 + tx] = tile[tx][dd];
  }
}

__global__ void k_init(float* __restrict__ u2, float* __restrict__ v2) {
  int i = blockIdx.x * 256 + threadIdx.x;
  u2[i] = 0.f; v2[i] = 0.f;
}

/* ---- cost-kernel GEMM: K2[i][j] = -max(|xi|^2+|yj|^2-2 xi.yj, 0)*log2e/reg
   one wave per 16x64 strip; software-pipelined: next A/B loads issued
   before the current WMMAs so matrix ops overlap VMEM ---- */
__global__ void k_gemm_cost(const _Float16* __restrict__ XPh, const _Float16* __restrict__ XQh,
                            const float* __restrict__ x2, const float* __restrict__ y2,
                            float* __restrict__ K2) {
  int wid  = blockIdx.x * 8 + (threadIdx.x >> 5);
  int lane = threadIdx.x & 31;
  int iTile = wid >> 7;          // 0..511
  int cg    = wid & 127;         // 0..127, 64 cols each
  int iBase = iTile * 16, jBase = cg * 64;
  int hl  = lane >> 4;           // lane-half select
  int l16 = lane & 15;
  const _Float16* Aptr = XPh + (size_t)(iBase + l16) * DIM + hl * 16;
  const _Float16* Bptr = XQh + (size_t)(jBase + l16) * DIM + hl * 16;
  v8f acc[4] = {};

  /* prologue: stage kk = 0 */
  v8h alo = *(const v8h*)(Aptr);
  v8h ahi = *(const v8h*)(Aptr + 8);
  v16h Bc[4];
#pragma unroll
  for (int t = 0; t < 4; t++)
    Bc[t] = *(const v16h*)(Bptr + (size_t)t * 16 * DIM);

#pragma unroll
  for (int kk = 0; kk < DIM - 32; kk += 32) {
    int kn = kk + 32;
    /* issue next-iteration loads first */
    v8h alo_n = *(const v8h*)(Aptr + kn);
    v8h ahi_n = *(const v8h*)(Aptr + kn + 8);
    v16h Bn[4];
#pragma unroll
    for (int t = 0; t < 4; t++)
      Bn[t] = *(const v16h*)(Bptr + (size_t)t * 16 * DIM + kn);
    /* WMMAs on data loaded one iteration ago */
    v16h A;
#pragma unroll
    for (int s = 0; s < 8; s++) { A[s] = alo[s]; A[s + 8] = ahi[s]; }
#pragma unroll
    for (int t = 0; t < 4; t++)
      acc[t] = __builtin_amdgcn_wmma_f32_16x16x32_f16(false, A, false, Bc[t],
                                                      (short)0, acc[t], false, false);
    alo = alo_n; ahi = ahi_n;
#pragma unroll
    for (int t = 0; t < 4; t++) Bc[t] = Bn[t];
  }
  /* epilogue: last K-slice */
  {
    v16h A;
#pragma unroll
    for (int s = 0; s < 8; s++) { A[s] = alo[s]; A[s + 8] = ahi[s]; }
#pragma unroll
    for (int t = 0; t < 4; t++)
      acc[t] = __builtin_amdgcn_wmma_f32_16x16x32_f16(false, A, false, Bc[t],
                                                      (short)0, acc[t], false, false);
  }

  int iOff = iBase + hl * 8;
#pragma unroll
  for (int t = 0; t < 4; t++) {
    int j = jBase + t * 16 + l16;
    float yj = y2[j];
#pragma unroll
    for (int r = 0; r < 8; r++) {
      int i = iOff + r;
      float d2 = fmaxf(x2[i] + yj - 2.0f * acc[t][r], 0.0f);
      K2[(size_t)i * N_PTS + j] = -d2 * KSCALE;
    }
  }
}

/* ---- u2[i] = -13 - log2( sum_j 2^(K2[i][j]+v2[j]) ), vectorized online LSE ---- */
__global__ void k_update_u(const float* __restrict__ K2, const float* __restrict__ v2,
                           float* __restrict__ u2) {
  int row = blockIdx.x, tid = threadIdx.x;
  const float* Kr = K2 + (size_t)row * N_PTS;
  float m = NEG_BIG, s = 0.f;
  for (int j = tid * 4; j < N_PTS; j += 1024) {
    float4 kx = *(const float4*)(Kr + j);
    float4 vv = *(const float4*)(v2 + j);
    lse4(m, s, kx.x + vv.x, kx.y + vv.y, kx.z + vv.z, kx.w + vv.w);
  }
  __shared__ float sm[256], ss[256];
  sm[tid] = m; ss[tid] = s; __syncthreads();
  for (int off = 128; off > 0; off >>= 1) {
    if (tid < off) {
      float m1 = sm[tid], s1 = ss[tid];
      float m2 = sm[tid + off], s2 = ss[tid + off];
      float M = fmaxf(m1, m2);
      sm[tid] = M;
      ss[tid] = s1 * fexp2(m1 - M) + s2 * fexp2(m2 - M);
    }
    __syncthreads();
  }
  if (tid == 0) u2[row] = -13.0f - (sm[0] + flog2(ss[0]));
}

/* ---- column LSE: 4 cols/thread, 4 rows/step, split over 8 row-groups ---- */
__global__ void k_update_v_part(const float* __restrict__ K2, const float* __restrict__ u2,
                                float* __restrict__ pm, float* __restrict__ ps) {
  int j0 = (blockIdx.x * 256 + threadIdx.x) * 4;
  int rbeg = blockIdx.y * 1024, rend = rbeg + 1024;
  float m0 = NEG_BIG, m1 = NEG_BIG, m2 = NEG_BIG, m3 = NEG_BIG;
  float s0 = 0.f, s1 = 0.f, s2 = 0.f, s3 = 0.f;
  for (int i = rbeg; i < rend; i += 4) {
    float4 r0 = *(const float4*)(K2 + (size_t)(i + 0) * N_PTS + j0);
    float4 r1 = *(const float4*)(K2 + (size_t)(i + 1) * N_PTS + j0);
    float4 r2 = *(const float4*)(K2 + (size_t)(i + 2) * N_PTS + j0);
    float4 r3 = *(const float4*)(K2 + (size_t)(i + 3) * N_PTS + j0);
    float ua = u2[i], ub = u2[i + 1], uc = u2[i + 2], ud = u2[i + 3];
    lse4(m0, s0, r0.x + ua, r1.x + ub, r2.x + uc, r3.x + ud);
    lse4(m1, s1, r0.y + ua, r1.y + ub, r2.y + uc, r3.y + ud);
    lse4(m2, s2, r0.z + ua, r1.z + ub, r2.z + uc, r3.z + ud);
    lse4(m3, s3, r0.w + ua, r1.w + ub, r2.w + uc, r3.w + ud);
  }
  size_t o = (size_t)blockIdx.y * N_PTS + j0;
  *(float4*)(pm + o) = make_float4(m0, m1, m2, m3);
  *(float4*)(ps + o) = make_float4(s0, s1, s2, s3);
}

__global__ void k_update_v_fin(const float* __restrict__ pm, const float* __restrict__ ps,
                               float* __restrict__ v2) {
  int j0 = (blockIdx.x * 256 + threadIdx.x) * 4;
  float m[4] = {NEG_BIG, NEG_BIG, NEG_BIG, NEG_BIG};
  float s[4] = {0.f, 0.f, 0.f, 0.f};
#pragma unroll
  for (int t = 0; t < 8; t++) {
    float4 mt = *(const float4*)(pm + (size_t)t * N_PTS + j0);
    float4 st = *(const float4*)(ps + (size_t)t * N_PTS + j0);
    float mv[4] = {mt.x, mt.y, mt.z, mt.w};
    float sv[4] = {st.x, st.y, st.z, st.w};
#pragma unroll
    for (int c = 0; c < 4; c++) {
      float M = fmaxf(m[c], mv[c]);
      s[c] = s[c] * fexp2(m[c] - M) + sv[c] * fexp2(mv[c] - M);
      m[c] = M;
    }
  }
  float4 out;
  out.x = -13.0f - (m[0] + flog2(s[0]));
  out.y = -13.0f - (m[1] + flog2(s[1]));
  out.z = -13.0f - (m[2] + flog2(s[2]));
  out.w = -13.0f - (m[3] + flog2(s[3]));
  *(float4*)(v2 + j0) = out;
}

/* ---- Out = (N * exp(K+u+v)) @ XQ ; A = 2^(K2+u2+v2+13) built on the fly.
   Software-pipelined: next K2/v2/B loads issued, then exp2 VALU work on the
   current slice hides them, then the WMMAs (data one iteration old) ---- */
__global__ void k_gemm_out(const float* __restrict__ K2, const float* __restrict__ u2,
                           const float* __restrict__ v2, const _Float16* __restrict__ XQt,
                           float* __restrict__ Out) {
  int wid  = blockIdx.x * 8 + (threadIdx.x >> 5);
  int lane = threadIdx.x & 31;
  int iBlock = wid >> 3;         // 0..511
  int cg     = wid & 7;          // 64 out cols each
  int iBase = iBlock * 16, dBase = cg * 64;
  int hl = lane >> 4, l16 = lane & 15;
  int arow  = iBase + l16;
  int abase = hl * 16, klo = hl * 16;
  float uval = u2[arow] + 13.0f;               // fold 1/p = N = 2^13
  const float* Kr = K2 + (size_t)arow * N_PTS;
  const _Float16* Bbase = XQt + (size_t)(dBase + l16) * N_PTS + klo;
  v8f acc[4] = {};

  float4 c[4], w[4];
  v16h B[4];
  /* prologue: stage kk = 0 */
  {
    int k0 = abase;
    c[0] = *(const float4*)(Kr + k0);
    c[1] = *(const float4*)(Kr + k0 + 4);
    c[2] = *(const float4*)(Kr + k0 + 8);
    c[3] = *(const float4*)(Kr + k0 + 12);
    w[0] = *(const float4*)(v2 + k0);
    w[1] = *(const float4*)(v2 + k0 + 4);
    w[2] = *(const float4*)(v2 + k0 + 8);
    w[3] = *(const float4*)(v2 + k0 + 12);
#pragma unroll
    for (int t = 0; t < 4; t++)
      B[t] = *(const v16h*)(Bbase + (size_t)t * 16 * N_PTS);
  }

  for (int kk = 0; kk < N_PTS - 32; kk += 32) {
    int kn = kk + 32 + abase;
    /* issue next-iteration loads */
    float4 cn[4], wn[4];
    v16h Bn[4];
    cn[0] = *(const float4*)(Kr + kn);
    cn[1] = *(const float4*)(Kr + kn + 4);
    cn[2] = *(const float4*)(Kr + kn + 8);
    cn[3] = *(const float4*)(Kr + kn + 12);
    wn[0] = *(const float4*)(v2 + kn);
    wn[1] = *(const float4*)(v2 + kn + 4);
    wn[2] = *(const float4*)(v2 + kn + 8);
    wn[3] = *(const float4*)(v2 + kn + 12);
#pragma unroll
    for (int t = 0; t < 4; t++)
      Bn[t] = *(const v16h*)(Bbase + (size_t)t * 16 * N_PTS + kk + 32);
    /* exp2 pipeline on current slice (hides the loads above) */
    float cv[16] = {c[0].x,c[0].y,c[0].z,c[0].w, c[1].x,c[1].y,c[1].z,c[1].w,
                    c[2].x,c[2].y,c[2].z,c[2].w, c[3].x,c[3].y,c[3].z,c[3].w};
    float wv[16] = {w[0].x,w[0].y,w[0].z,w[0].w, w[1].x,w[1].y,w[1].z,w[1].w,
                    w[2].x,w[2].y,w[2].z,w[2].w, w[3].x,w[3].y,w[3].z,w[3].w};
    v16h A;
#pragma unroll
    for (int s = 0; s < 16; s++) A[s] = (_Float16)fexp2(cv[s] + wv[s] + uval);
#pragma unroll
    for (int t = 0; t < 4; t++)
      acc[t] = __builtin_amdgcn_wmma_f32_16x16x32_f16(false, A, false, B[t],
                                                      (short)0, acc[t], false, false);
#pragma unroll
    for (int q = 0; q < 4; q++) { c[q] = cn[q]; w[q] = wn[q]; B[q] = Bn[q]; }
  }
  /* epilogue: last K-slice */
  {
    float cv[16] = {c[0].x,c[0].y,c[0].z,c[0].w, c[1].x,c[1].y,c[1].z,c[1].w,
                    c[2].x,c[2].y,c[2].z,c[2].w, c[3].x,c[3].y,c[3].z,c[3].w};
    float wv[16] = {w[0].x,w[0].y,w[0].z,w[0].w, w[1].x,w[1].y,w[1].z,w[1].w,
                    w[2].x,w[2].y,w[2].z,w[2].w, w[3].x,w[3].y,w[3].z,w[3].w};
    v16h A;
#pragma unroll
    for (int s = 0; s < 16; s++) A[s] = (_Float16)fexp2(cv[s] + wv[s] + uval);
#pragma unroll
    for (int t = 0; t < 4; t++)
      acc[t] = __builtin_amdgcn_wmma_f32_16x16x32_f16(false, A, false, B[t],
                                                      (short)0, acc[t], false, false);
  }

  int iOff = iBase + hl * 8;
#pragma unroll
  for (int t = 0; t < 4; t++)
#pragma unroll
    for (int r = 0; r < 8; r++)
      Out[(size_t)(iOff + r) * DIM + dBase + t * 16 + l16] = acc[t][r];
}

extern "C" void kernel_launch(void* const* d_in, const int* in_sizes, int n_in,
                              void* d_out, int out_size, void* d_ws, size_t ws_size,
                              hipStream_t stream) {
  (void)in_sizes; (void)n_in; (void)out_size; (void)ws_size;
  const float* XP = (const float*)d_in[0];
  const float* XQ = (const float*)d_in[1];
  float* Out = (float*)d_out;

  char* ws = (char*)d_ws;
  size_t off = 0;
  float* K2 = (float*)(ws + off);        off += (size_t)N_PTS * N_PTS * sizeof(float);  // 256 MB
  _Float16* XPh = (_Float16*)(ws + off); off += (size_t)N_PTS * DIM * sizeof(_Float16);
  _Float16* XQh = (_Float16*)(ws + off); off += (size_t)N_PTS * DIM * sizeof(_Float16);
  _Float16* XQt = (_Float16*)(ws + off); off += (size_t)N_PTS * DIM * sizeof(_Float16);
  float* x2 = (float*)(ws + off);        off += (size_t)N_PTS * sizeof(float);
  float* y2 = (float*)(ws + off);        off += (size_t)N_PTS * sizeof(float);
  float* u2 = (float*)(ws + off);        off += (size_t)N_PTS * sizeof(float);
  float* v2 = (float*)(ws + off);        off += (size_t)N_PTS * sizeof(float);
  float* pm = (float*)(ws + off);        off += (size_t)8 * N_PTS * sizeof(float);
  float* ps = (float*)(ws + off);        off += (size_t)8 * N_PTS * sizeof(float);

  k_prep<<<N_PTS, 256, 0, stream>>>(XP, XPh, x2);
  k_prep<<<N_PTS, 256, 0, stream>>>(XQ, XQh, y2);
  k_transpose<<<dim3(N_PTS / 32, DIM / 32), 256, 0, stream>>>(XQh, XQt);
  k_init<<<N_PTS / 256, 256, 0, stream>>>(u2, v2);
  k_gemm_cost<<<8192, 256, 0, stream>>>(XPh, XQh, x2, y2, K2);

  for (int it = 0; it < 50; it++) {
    k_update_u<<<N_PTS, 256, 0, stream>>>(K2, v2, u2);
    k_update_v_part<<<dim3(8, 8), 256, 0, stream>>>(K2, u2, pm, ps);
    k_update_v_fin<<<8, 256, 0, stream>>>(pm, ps, v2);
  }

  k_gemm_out<<<512, 256, 0, stream>>>(K2, u2, v2, XQt, Out);
}
